// MultiheadAttention_6648609374712
// MI455X (gfx1250) — matrix operand order repaired and checked by reading, not compiled, not run on hardware
//
#include <hip/hip_runtime.h>
#include <hip/hip_bf16.h>

typedef __attribute__((ext_vector_type(16))) __bf16 v16bf;
typedef __attribute__((ext_vector_type(8)))  float  v8f;

namespace {
constexpr int S_LEN = 4096;
constexpr int EMB   = 1024;
constexpr int NHEAD = 16;
constexpr int HD    = 64;
constexpr int QKVN  = 3 * EMB;
constexpr int PSTRIDE = 56;  // P-tile LDS row stride (elements): 112B = 28 banks -> conflict-free b128 reads
}

// ---------- fragment helpers ----------

static __device__ __forceinline__ v8f wmma_bf16(const v16bf a, const v16bf b, const v8f c) {
  // 8 args: (neg_a, A, neg_b, B, c_mod, C, reuse_a, reuse_b)
  return __builtin_amdgcn_wmma_f32_16x16x32_bf16(false, a, false, b, (short)0, c, false, false);
}

// Load 8 consecutive bf16 (16 bytes) into fragment slots [bi..bi+7].
static __device__ __forceinline__ void load8_bf16(v16bf& a, int bi, const __bf16* p) {
  union { uint4 u; __bf16 h[8]; } cv;
  cv.u = *reinterpret_cast<const uint4*>(p);
#pragma unroll
  for (int j = 0; j < 8; ++j) a[bi + j] = cv.h[j];
}

// ---------- kernel 0: one-shot f32 -> bf16 conversion prepass ----------
// Pulls ALL conversion work out of the GEMM hot loops (x is re-read 12x, W 256x otherwise).

__global__ void __launch_bounds__(256) f32_to_bf16_kernel(
    const float* __restrict__ src, __bf16* __restrict__ dst, int n) {
  const int i = (blockIdx.x * 256 + threadIdx.x) * 8;
  if (i >= n) return;
  const float4 f0 = *reinterpret_cast<const float4*>(src + i);
  const float4 f1 = *reinterpret_cast<const float4*>(src + i + 4);
  union { uint4 u; __bf16 h[8]; } cv;
  cv.h[0] = (__bf16)f0.x; cv.h[1] = (__bf16)f0.y;
  cv.h[2] = (__bf16)f0.z; cv.h[3] = (__bf16)f0.w;
  cv.h[4] = (__bf16)f1.x; cv.h[5] = (__bf16)f1.y;
  cv.h[6] = (__bf16)f1.z; cv.h[7] = (__bf16)f1.w;
  *reinterpret_cast<uint4*>(dst + i) = cv.u;
}

// ---------- kernel 1: qkv = x @ Wqkv^T + bqkv, scattered into per-head q/k/v bf16 ----------
// Block = 128 threads (4 waves). Block tile: 16(M) x 256(N); wave tile 16x64 (4 WMMA accs).
// All operands pre-converted bf16 -> pure load+WMMA inner loop (no VALU).
// Q,K stored [H][S][HD]; V stored TRANSPOSED [H][HD][S] so the PV B-fragment load is contiguous.

__global__ void __launch_bounds__(128) qkv_proj_kernel(
    const __bf16* __restrict__ Xb, const __bf16* __restrict__ Wb,
    const float* __restrict__ bqkv,
    __bf16* __restrict__ Qb, __bf16* __restrict__ Kb, __bf16* __restrict__ Vt) {
  const int lane = threadIdx.x & 31;
  const int wave = threadIdx.x >> 5;
  const int col  = lane & 15;
  const int hsel = lane >> 4;          // 0: lanes 0-15, 1: lanes 16-31
  const int m0   = blockIdx.y * 16;
  const int n0   = blockIdx.x * 256 + wave * 64;
  const int kbA  = hsel * 16;          // A-fragment K half
  const int kbB  = hsel * 16;          // B-fragment K half

  v8f acc[4] = {};
  const __bf16* xr = Xb + (size_t)(m0 + col) * EMB;

  for (int k0 = 0; k0 < EMB; k0 += 32) {
    v16bf a;
    load8_bf16(a, 0, xr + k0 + kbA);
    load8_bf16(a, 8, xr + k0 + 8 + kbA);
#pragma unroll
    for (int t = 0; t < 4; ++t) {
      v16bf b;  // B[k][n] = Wqkv[n][k] (row-major, contiguous in k)
      const __bf16* wr = Wb + (size_t)(n0 + t * 16 + col) * EMB + k0 + kbB;
      load8_bf16(b, 0, wr);
      load8_bf16(b, 8, wr + 8);
      acc[t] = wmma_bf16(a, b, acc[t]);
    }
  }

#pragma unroll
  for (int t = 0; t < 4; ++t) {
    const int n    = n0 + t * 16 + col;
    const int head = n / (3 * HD);
    const int j    = n % (3 * HD);
    const int part = j / HD;
    const int d    = j % HD;
    const float bias = bqkv[n];
    // q/k: [H][S][HD] (row stride HD); v: [H][HD][S] (row stride 1)
    __bf16* dst;
    size_t rstride;
    if (part == 0)      { dst = Qb + (size_t)head * S_LEN * HD + d; rstride = HD; }
    else if (part == 1) { dst = Kb + (size_t)head * S_LEN * HD + d; rstride = HD; }
    else                { dst = Vt + ((size_t)head * HD + d) * S_LEN; rstride = 1; }
#pragma unroll
    for (int r = 0; r < 8; ++r) {
      const int row = m0 + r + hsel * 8;
      dst[(size_t)row * rstride] = (__bf16)(acc[t][r] + bias);
    }
  }
}

// ---------- kernel 2: flash attention, 1 wave per (head, 16-query tile) ----------

__global__ void __launch_bounds__(128) flash_attn_kernel(
    const __bf16* __restrict__ Q, const __bf16* __restrict__ K,
    const __bf16* __restrict__ Vt, __bf16* __restrict__ OUT) {
  __shared__ __align__(16) __bf16 pbuf[4][16 * PSTRIDE];  // per-wave P staging (C->A relayout)

  const int lane = threadIdx.x & 31;
  const int wave = threadIdx.x >> 5;
  const int col  = lane & 15;
  const int hsel = lane >> 4;
  const int head = blockIdx.y;
  const int q0   = (blockIdx.x * 4 + wave) * 16;
  const int kbA  = hsel * 16;
  const int kbB  = hsel * 16;

  const __bf16* Qh  = Q  + (size_t)head * S_LEN * HD;
  const __bf16* Kh  = K  + (size_t)head * S_LEN * HD;
  const __bf16* Vth = Vt + (size_t)head * HD * S_LEN;   // [HD][S]

  // Q fragments, A layout: M = col, d split into chunks [0,32) and [32,64)
  v16bf qa0, qa1;
  {
    const __bf16* qr = Qh + (size_t)(q0 + col) * HD;
    load8_bf16(qa0, 0, qr + kbA);
    load8_bf16(qa0, 8, qr + 8 + kbA);
    load8_bf16(qa1, 0, qr + 32 + kbA);
    load8_bf16(qa1, 8, qr + 40 + kbA);
  }

  v8f o[4] = {};
  float m_run[8], l_run[8];
#pragma unroll
  for (int r = 0; r < 8; ++r) { m_run[r] = -__builtin_inff(); l_run[r] = 0.0f; }

  constexpr float scale = 0.125f;  // 1/sqrt(64)

  for (int key0 = 0; key0 < S_LEN; key0 += 32) {
    if (key0 + 32 < S_LEN) {  // prefetch next K/V tiles (global_prefetch)
      __builtin_prefetch(Kh + (size_t)(key0 + 32 + lane) * HD, 0, 0);
      __builtin_prefetch(Vth + (size_t)lane * S_LEN + key0 + 32, 0, 0);
    }

    // S = Q * K^T for 32 keys -> two 16x16 f32 tiles (contract over d=64 in 2 chunks)
    v8f s0 = {}, s1 = {};
    {
      v16bf bk;  // B[d][key] = Kh[key][d] (contiguous in d)
      const __bf16* kr0 = Kh + (size_t)(key0 + col) * HD;
      load8_bf16(bk, 0, kr0 + kbB);
      load8_bf16(bk, 8, kr0 + kbB + 8);
      s0 = wmma_bf16(qa0, bk, s0);
      load8_bf16(bk, 0, kr0 + 32 + kbB);
      load8_bf16(bk, 8, kr0 + 40 + kbB);
      s0 = wmma_bf16(qa1, bk, s0);
      const __bf16* kr1 = kr0 + 16 * HD;
      load8_bf16(bk, 0, kr1 + kbB);
      load8_bf16(bk, 8, kr1 + kbB + 8);
      s1 = wmma_bf16(qa0, bk, s1);
      load8_bf16(bk, 0, kr1 + 32 + kbB);
      load8_bf16(bk, 8, kr1 + 40 + kbB);
      s1 = wmma_bf16(qa1, bk, s1);
    }

    // Online softmax. C layout: element r of each lane is row (r + 8*hsel), col (lane&15).
    float mx[8];
#pragma unroll
    for (int r = 0; r < 8; ++r) {
      s0[r] *= scale; s1[r] *= scale;
      mx[r] = fmaxf(s0[r], s1[r]);
    }
#pragma unroll
    for (int off = 1; off < 16; off <<= 1)
#pragma unroll
      for (int r = 0; r < 8; ++r)
        mx[r] = fmaxf(mx[r], __shfl_xor(mx[r], off, 32));

    float alpha[8], ls[8];
#pragma unroll
    for (int r = 0; r < 8; ++r) {
      const float mnew = fmaxf(m_run[r], mx[r]);
      alpha[r] = __expf(m_run[r] - mnew);
      m_run[r] = mnew;
      s0[r] = __expf(s0[r] - mnew);
      s1[r] = __expf(s1[r] - mnew);
      ls[r] = s0[r] + s1[r];
    }
#pragma unroll
    for (int off = 1; off < 16; off <<= 1)
#pragma unroll
      for (int r = 0; r < 8; ++r)
        ls[r] += __shfl_xor(ls[r], off, 32);
#pragma unroll
    for (int r = 0; r < 8; ++r)
      l_run[r] = l_run[r] * alpha[r] + ls[r];
#pragma unroll
    for (int t = 0; t < 4; ++t)
#pragma unroll
      for (int r = 0; r < 8; ++r)
        o[t][r] *= alpha[r];

    // Relayout P (C layout) -> A fragment via per-wave LDS tile (padded stride: no bank conflicts).
    __bf16* pb = pbuf[wave];
#pragma unroll
    for (int r = 0; r < 8; ++r) {
      const int row = r + hsel * 8;
      pb[row * PSTRIDE + col]      = (__bf16)s0[r];
      pb[row * PSTRIDE + col + 16] = (__bf16)s1[r];
    }
    asm volatile("s_wait_dscnt 0" ::: "memory");  // same-wave LDS RAW fence
    v16bf pa;
    {
      const __bf16* pr = pb + col * PSTRIDE;
      load8_bf16(pa, 0, pr + kbA);
      load8_bf16(pa, 8, pr + 8 + kbA);
    }

    // O += P * V : contract over 32 keys, 4 d-tiles of 16.
    // V is transposed [HD][S]: B[key][d] = Vth[d][key] -> contiguous along keys.
#pragma unroll
    for (int t = 0; t < 4; ++t) {
      v16bf bv;
      const __bf16* vp = Vth + (size_t)(t * 16 + col) * S_LEN + key0 + kbB;
      load8_bf16(bv, 0, vp);
      load8_bf16(bv, 8, vp + 8);
      o[t] = wmma_bf16(pa, bv, o[t]);
    }
  }

  float inv[8];
#pragma unroll
  for (int r = 0; r < 8; ++r) inv[r] = 1.0f / l_run[r];
#pragma unroll
  for (int t = 0; t < 4; ++t)
#pragma unroll
    for (int r = 0; r < 8; ++r) {
      const int row = q0 + r + hsel * 8;
      const int cg  = head * HD + t * 16 + col;
      OUT[(size_t)row * EMB + cg] = (__bf16)(o[t][r] * inv[r]);
    }
}

// ---------- kernel 3: out = values @ Wo^T + bo (f32 output) ----------

__global__ void __launch_bounds__(128) out_proj_kernel(
    const __bf16* __restrict__ Vals, const __bf16* __restrict__ Wob,
    const float* __restrict__ bo, float* __restrict__ Out) {
  const int lane = threadIdx.x & 31;
  const int wave = threadIdx.x >> 5;
  const int col  = lane & 15;
  const int hsel = lane >> 4;
  const int m0   = blockIdx.y * 16;
  const int n0   = blockIdx.x * 256 + wave * 64;
  const int kbA  = hsel * 16;
  const int kbB  = hsel * 16;

  v8f acc[4] = {};
  const __bf16* vr = Vals + (size_t)(m0 + col) * EMB;

  for (int k0 = 0; k0 < EMB; k0 += 32) {
    v16bf a;
    load8_bf16(a, 0, vr + k0 + kbA);
    load8_bf16(a, 8, vr + k0 + 8 + kbA);
#pragma unroll
    for (int t = 0; t < 4; ++t) {
      v16bf b;
      const __bf16* wr = Wob + (size_t)(n0 + t * 16 + col) * EMB + k0 + kbB;
      load8_bf16(b, 0, wr);
      load8_bf16(b, 8, wr + 8);
      acc[t] = wmma_bf16(a, b, acc[t]);
    }
  }

#pragma unroll
  for (int t = 0; t < 4; ++t) {
    const int n = n0 + t * 16 + col;
    const float bias = bo[n];
#pragma unroll
    for (int r = 0; r < 8; ++r) {
      const int row = m0 + r + hsel * 8;
      Out[(size_t)row * EMB + n] = acc[t][r] + bias;
    }
  }
}

// ---------- launch ----------

extern "C" void kernel_launch(void* const* d_in, const int* in_sizes, int n_in,
                              void* d_out, int out_size, void* d_ws, size_t ws_size,
                              hipStream_t stream) {
  (void)in_sizes; (void)n_in; (void)out_size; (void)ws_size;
  const float* x    = (const float*)d_in[0];
  const float* Wqkv = (const float*)d_in[1];
  const float* bqkv = (const float*)d_in[2];
  const float* Wo   = (const float*)d_in[3];
  const float* bo   = (const float*)d_in[4];
  float* out = (float*)d_out;

  // Workspace layout (bf16 elements):
  //   Qb/Kb [H,S,HD], Vt [H,HD,S], Valb [S,E]  : 4 x 4Mi elems (8 MB each)
  //   Xb [S,E] : 4Mi, Wqkvb [3E,E] : 3Mi, Wob [E,E] : 1Mi   -> total 48 MB
  const size_t perBuf = (size_t)NHEAD * S_LEN * HD;   // 4Mi
  __bf16* Qb    = (__bf16*)d_ws;
  __bf16* Kb    = Qb + perBuf;
  __bf16* Vt    = Kb + perBuf;
  __bf16* Valb  = Vt + perBuf;
  __bf16* Xb    = Valb + perBuf;
  __bf16* Wqkvb = Xb + (size_t)S_LEN * EMB;
  __bf16* Wob   = Wqkvb + (size_t)QKVN * EMB;

  // Prepass: one-shot f32 -> bf16 (bandwidth-trivial, removes all cvt from GEMM hot loops).
  const int nX = S_LEN * EMB, nWq = QKVN * EMB, nWo = EMB * EMB;
  f32_to_bf16_kernel<<<nX  / 2048, 256, 0, stream>>>(x,    Xb,    nX);
  f32_to_bf16_kernel<<<nWq / 2048, 256, 0, stream>>>(Wqkv, Wqkvb, nWq);
  f32_to_bf16_kernel<<<nWo / 2048, 256, 0, stream>>>(Wo,   Wob,   nWo);

  dim3 g1(QKVN / 256, S_LEN / 16);   // (12, 256)
  qkv_proj_kernel<<<g1, 128, 0, stream>>>(Xb, Wqkvb, bqkv, Qb, Kb, Vt);

  dim3 g2(S_LEN / 64, NHEAD);        // (64, 16)
  flash_attn_kernel<<<g2, 128, 0, stream>>>(Qb, Kb, Vt, Valb);

  dim3 g3(EMB / 256, S_LEN / 16);    // (4, 256)
  out_proj_kernel<<<g3, 128, 0, stream>>>(Valb, Wob, bo, out);
}
